// ColumnarTransformerBlock_21947282882792
// MI455X (gfx1250) — matrix operand order repaired and checked by reading, not compiled, not run on hardware
//
#include <hip/hip_runtime.h>
#include <hip/hip_bf16.h>

#define BB 32
#define TT 512
#define DD 1024
#define II 768
#define HH 16
#define HD 64
#define EE 8
#define BT (BB * TT)   // 16384

typedef __attribute__((ext_vector_type(16))) __bf16 v16bf;
typedef __attribute__((ext_vector_type(8)))  float  v8f;
typedef __attribute__((ext_vector_type(4))) unsigned int u32x4;
typedef __attribute__((ext_vector_type(8))) int i32x8;
typedef __attribute__((ext_vector_type(4))) int i32x4;

#if __has_builtin(__builtin_amdgcn_tensor_load_to_lds)
#define HAS_TDM 1
#else
#define HAS_TDM 0
#endif

__device__ __forceinline__ unsigned short f2bfbits(float f) {
    unsigned u = __builtin_bit_cast(unsigned, f);
    u += 0x7FFFu + ((u >> 16) & 1u);       // round-to-nearest-even bf16
    return (unsigned short)(u >> 16);
}
__device__ __forceinline__ __bf16 bits2bf(unsigned short h) {
    return __builtin_bit_cast(__bf16, h);
}

#if HAS_TDM
// Issue a TDM copy of a 64-row x 32-col bf16 tile from a row-major [rows x K]
// matrix at (row0, k0) into LDS byte offset lds_off. Uniform (SGPR) operands.
__device__ __forceinline__ void tdm_issue_tile(const unsigned short* base,
                                               unsigned lds_off, int rows,
                                               int K, int row0, int k0) {
    unsigned long long ga = (unsigned long long)(size_t)base +
        ((unsigned long long)(unsigned)row0 * (unsigned)K + (unsigned)k0) * 2ull;
    u32x4 g0;
    g0[0] = 1u;                                                 // count=1, user mode
    g0[1] = lds_off;                                            // LDS byte address
    g0[2] = (unsigned)(ga & 0xFFFFFFFFull);
    g0[3] = (unsigned)((ga >> 32) & 0x01FFFFFFull) | (2u << 30);// type=2 (image)
    i32x8 g1;
    g1[0] = (int)(1u << 16);                                    // data_size = 2 bytes
    g1[1] = (int)(((unsigned)K & 0xFFFFu) << 16);               // tensor_dim0 lo16
    g1[2] = (int)((((unsigned)rows & 0xFFFFu) << 16) | ((unsigned)K >> 16));
    g1[3] = (int)((32u << 16) | ((unsigned)rows >> 16));        // tile_dim0=32
    g1[4] = 64;                                                 // tile_dim1=64
    g1[5] = K;                                                  // dim0_stride lo32
    g1[6] = 0;
    g1[7] = 0;
    i32x4 gz; gz[0] = 0; gz[1] = 0; gz[2] = 0; gz[3] = 0;
#if defined(__clang_major__) && __clang_major__ >= 23
    i32x8 gz8;
    for (int q = 0; q < 8; ++q) gz8[q] = 0;
    __builtin_amdgcn_tensor_load_to_lds(g0, g1, gz, gz, gz8, 0);
#else
    __builtin_amdgcn_tensor_load_to_lds(g0, g1, gz, gz, 0);
#endif
}
#endif

// ---------------------------------------------------------------------------
// fp32 -> bf16 bulk convert (with prefetch)
// ---------------------------------------------------------------------------
__global__ __launch_bounds__(256)
void f32_to_bf16_kernel(const float* __restrict__ src,
                        unsigned short* __restrict__ dst, long long n) {
    long long i = (long long)blockIdx.x * 256 + threadIdx.x;
    if (i + 8192 < n) __builtin_prefetch(src + i + 8192, 0, 1);
    if (i < n) dst[i] = f2bfbits(src[i]);
}

// ---------------------------------------------------------------------------
// Router: pooled mean -> logits -> top-2 softmax expert weights
// ---------------------------------------------------------------------------
__global__ __launch_bounds__(256)
void router_kernel(const float* __restrict__ h, const float* __restrict__ rw,
                   const float* __restrict__ temp_p,
                   float* __restrict__ logits_out, float* __restrict__ expert_w) {
    __shared__ float pooled[DD];
    __shared__ float lg[EE];
    const int b = blockIdx.x;
    for (int d = threadIdx.x; d < DD; d += 256) {
        float s = 0.f;
        const float* p = h + (size_t)b * TT * DD + d;
        for (int t = 0; t < TT; ++t) s += p[(size_t)t * DD];
        pooled[d] = s * (1.f / TT);
    }
    __syncthreads();
    float temp = fminf(fmaxf(temp_p[0], 0.1f), 10.f);
    if (threadIdx.x < EE) {
        float s = 0.f;
        const float* w = rw + (size_t)threadIdx.x * DD;
        for (int d = 0; d < DD; ++d) s += pooled[d] * w[d];
        s /= temp;
        lg[threadIdx.x] = s;
        logits_out[b * EE + threadIdx.x] = s;
    }
    __syncthreads();
    if (threadIdx.x == 0) {
        int i0 = 0;
        for (int e = 1; e < EE; ++e) if (lg[e] > lg[i0]) i0 = e;
        int i1 = -1;
        for (int e = 0; e < EE; ++e) {
            if (e == i0) continue;
            if (i1 < 0 || lg[e] > lg[i1]) i1 = e;
        }
        float e1 = __expf(lg[i1] - lg[i0]);
        float inv = 1.f / (1.f + e1);
        for (int e = 0; e < EE; ++e) expert_w[b * EE + e] = 0.f;
        expert_w[b * EE + i0] = inv;
        expert_w[b * EE + i1] = e1 * inv;
    }
}

// ---------------------------------------------------------------------------
// All-bf16 tiled WMMA GEMM: C[M,N] = A[M,K] * W[N,K]^T
// Both A and W tiles staged by the Tensor Data Mover, double-buffered:
// dynamic LDS 16KB: Ws0 @0 | Ws1 @4096 | As0 @8192 | As1 @12288 (bytes)
// Pipeline: issue next pair -> s_wait_tensorcnt 2 -> barrier -> 4x WMMA.
// block = 128 (4 waves), block tile 64x64, K step 32, grid = (M/64, N/64)
// ---------------------------------------------------------------------------
__global__ __launch_bounds__(128)
void gemm_bf16_wmma(const unsigned short* __restrict__ Abf,
                    const unsigned short* __restrict__ Wbf,
                    float* __restrict__ C, int M, int N, int K) {
    extern __shared__ char smem[];
    unsigned short* lds = (unsigned short*)smem;
    const int tid  = threadIdx.x;
    const int lane = tid & 31;
    const int wave = tid >> 5;
    const int m0 = blockIdx.x * 64;
    const int n0 = blockIdx.y * 64;
    const int rowM = lane & 15;
    const int kgrp = lane >> 4;
    const int nk = K / 32;

    v8f acc[4];
    for (int i = 0; i < 4; ++i)
        for (int j = 0; j < 8; ++j) acc[i][j] = 0.f;

#if HAS_TDM
    if (wave == 0) {                       // prologue: stage tile 0
        tdm_issue_tile(Wbf, 0u,     N, K, n0, 0);
        tdm_issue_tile(Abf, 8192u,  M, K, m0, 0);
    }
#endif
    for (int kt = 0; kt < nk; ++kt) {
        const unsigned cur = (unsigned)(kt & 1);
#if HAS_TDM
        if (wave == 0) {
            if (kt + 1 < nk) {             // stage next tile while current lands
                unsigned nb = (unsigned)((kt + 1) & 1);
                tdm_issue_tile(Wbf, nb * 4096u,         N, K, n0, (kt + 1) * 32);
                tdm_issue_tile(Abf, 8192u + nb * 4096u, M, K, m0, (kt + 1) * 32);
                __builtin_amdgcn_s_wait_tensorcnt(2);   // current pair complete
            } else {
                __builtin_amdgcn_s_wait_tensorcnt(0);
            }
        }
#else
        for (int i = tid; i < 64 * 32; i += 128) {
            int r = i >> 5, c = i & 31;
            lds[cur * 2048u + i]         = Wbf[(size_t)(n0 + r) * K + kt * 32 + c];
            lds[4096u + cur * 2048u + i] = Abf[(size_t)(m0 + r) * K + kt * 32 + c];
        }
#endif
        __syncthreads();
        const unsigned short* Wc = lds + cur * 2048u;           // elements
        const unsigned short* Ac = lds + 4096u + cur * 2048u;
        v16bf af;
        for (int i = 0; i < 16; ++i) {
            int kk = (i < 8) ? (kgrp * 8 + i) : (16 + kgrp * 8 + (i - 8));
            af[i] = bits2bf(Ac[(wave * 16 + rowM) * 32 + kk]);
        }
        for (int nt = 0; nt < 4; ++nt) {
            v16bf bfv;
            for (int i = 0; i < 16; ++i)
                bfv[i] = bits2bf(Wc[(nt * 16 + rowM) * 32 + ((i < 8) ? (kgrp * 8 + i) : (16 + kgrp * 8 + (i - 8)))]);
            acc[nt] = __builtin_amdgcn_wmma_f32_16x16x32_bf16(
                false, af, false, bfv, (short)0, acc[nt], false, false);
        }
        __syncthreads();   // all reads done before buffer reused two steps later
    }
    for (int nt = 0; nt < 4; ++nt)
        for (int r = 0; r < 8; ++r) {
            int mm = m0 + wave * 16 + r + 8 * kgrp;
            int nn = n0 + nt * 16 + rowM;
            C[(size_t)mm * N + nn] = acc[nt][r];
        }
}

// ---------------------------------------------------------------------------
// RoPE in-place on q (c=0) and k (c=1); one thread per (d, d+32) pair.
// ---------------------------------------------------------------------------
__global__ __launch_bounds__(256)
void rope_kernel(float* __restrict__ qkv, const float* __restrict__ cosT,
                 const float* __restrict__ sinT) {
    long long idx = (long long)blockIdx.x * 256 + threadIdx.x;
    int d = (int)(idx & 31);
    int h = (int)((idx >> 5) & 15);
    int c = (int)((idx >> 9) & 1);
    long long bt = idx >> 10;
    int t = (int)(bt & (TT - 1));
    long long base = bt * (3 * DD) + (long long)c * DD + (long long)h * HD;
    float x1 = qkv[base + d];
    float x2 = qkv[base + d + 32];
    float c1 = cosT[t * HD + d],      s1 = sinT[t * HD + d];
    float c2 = cosT[t * HD + d + 32], s2 = sinT[t * HD + d + 32];
    qkv[base + d]      = x1 * c1 - x2 * s1;
    qkv[base + d + 32] = x2 * c2 + x1 * s2;
}

// ---------------------------------------------------------------------------
// Attention: block = 128 (4 waves) handles 64 query rows of one (b,h).
// Q/K/V staged through LDS as bf16, WMMA for QK^T and PV. Output written bf16
// (it is only consumed as the A-matrix of the o-projection GEMM).
// dynamic LDS: Qs 8KB | Ks 2KB | Vs 4KB | scores 64x512 f32 128KB
// ---------------------------------------------------------------------------
__global__ __launch_bounds__(128)
void attention_kernel(const float* __restrict__ qkv, unsigned short* __restrict__ attnb) {
    extern __shared__ char smem[];
    unsigned short* Qs = (unsigned short*)smem;              // [64][64]
    unsigned short* Ks = (unsigned short*)(smem + 8192);     // [16][64]
    unsigned short* Vs = (unsigned short*)(smem + 10240);    // [32][64]
    float* sc          = (float*)(smem + 14336);             // [64][TT]
    const int tid = threadIdx.x, lane = tid & 31, wave = tid >> 5;
    const int q0 = blockIdx.x * 64, h = blockIdx.y, b = blockIdx.z;
    const int rowM = lane & 15, kgrp = lane >> 4;
    const size_t rs = 3 * DD;

    for (int i = tid; i < 64 * 64; i += 128) {
        int r = i >> 6, c = i & 63;
        Qs[i] = f2bfbits(qkv[(size_t)(b * TT + q0 + r) * rs + h * HD + c]);
    }
    __syncthreads();
    v16bf qf[2];
    for (int f = 0; f < 2; ++f)
        for (int i = 0; i < 16; ++i) {
            int kk = f * 32 + ((i < 8) ? (kgrp * 8 + i) : (16 + kgrp * 8 + (i - 8)));
            qf[f][i] = bits2bf(Qs[(wave * 16 + rowM) * 64 + kk]);
        }

    for (int kt = 0; kt < TT / 16; ++kt) {
        for (int i = tid; i < 16 * 64; i += 128) {
            int r = i >> 6, c = i & 63;
            Ks[i] = f2bfbits(qkv[(size_t)(b * TT + kt * 16 + r) * rs + DD + h * HD + c]);
        }
        __syncthreads();
        v8f c8;
        for (int j = 0; j < 8; ++j) c8[j] = 0.f;
        for (int f = 0; f < 2; ++f) {
            v16bf kf;
            for (int i = 0; i < 16; ++i)
                kf[i] = bits2bf(Ks[rowM * 64 + f * 32 + ((i < 8) ? (kgrp * 8 + i) : (16 + kgrp * 8 + (i - 8)))]);
            c8 = __builtin_amdgcn_wmma_f32_16x16x32_bf16(
                false, qf[f], false, kf, (short)0, c8, false, false);
        }
        for (int r = 0; r < 8; ++r)
            sc[(wave * 16 + r + 8 * kgrp) * TT + kt * 16 + rowM] = c8[r] * 0.125f;
        __syncthreads();
    }

    if (tid < 64) {
        float* row = sc + (size_t)tid * TT;
        float m = -1e30f;
        for (int j = 0; j < TT; ++j) m = fmaxf(m, row[j]);
        float s = 0.f;
        for (int j = 0; j < TT; ++j) { float e = __expf(row[j] - m); row[j] = e; s += e; }
        float inv = 1.f / s;
        for (int j = 0; j < TT; ++j) row[j] *= inv;
    }
    __syncthreads();

    v8f o[4];
    for (int nt = 0; nt < 4; ++nt)
        for (int j = 0; j < 8; ++j) o[nt][j] = 0.f;
    for (int kt2 = 0; kt2 < TT / 32; ++kt2) {
        for (int i = tid; i < 32 * 64; i += 128) {
            int r = i >> 6, c = i & 63;
            Vs[i] = f2bfbits(qkv[(size_t)(b * TT + kt2 * 32 + r) * rs + 2 * DD + h * HD + c]);
        }
        __syncthreads();
        v16bf pa;
        for (int i = 0; i < 16; ++i) {
            int kk = (i < 8) ? (kgrp * 8 + i) : (16 + kgrp * 8 + (i - 8));
            pa[i] = bits2bf(f2bfbits(sc[(wave * 16 + rowM) * TT + kt2 * 32 + kk]));
        }
        for (int nt = 0; nt < 4; ++nt) {
            v16bf vf;
            for (int i = 0; i < 16; ++i)
                vf[i] = bits2bf(Vs[((i < 8) ? (kgrp * 8 + i) : (16 + kgrp * 8 + (i - 8))) * 64 + nt * 16 + rowM]);
            o[nt] = __builtin_amdgcn_wmma_f32_16x16x32_bf16(
                false, pa, false, vf, (short)0, o[nt], false, false);
        }
        __syncthreads();
    }
    for (int nt = 0; nt < 4; ++nt)
        for (int r = 0; r < 8; ++r) {
            size_t orow = (size_t)(b * TT + q0 + wave * 16 + r + 8 * kgrp);
            attnb[orow * DD + h * HD + nt * 16 + rowM] = f2bfbits(o[nt][r]);
        }
}

// ---------------------------------------------------------------------------
// outf = rms_norm(x + y), plus bf16 copy for the next GEMM's A matrix
// ---------------------------------------------------------------------------
__global__ __launch_bounds__(256)
void rmsnorm_residual(const float* __restrict__ x, const float* __restrict__ y,
                      float* __restrict__ outf, unsigned short* __restrict__ outb) {
    __shared__ float red[256];
    const size_t row = blockIdx.x;
    const float* xr = x + row * DD;
    const float* yr = y + row * DD;
    float ss = 0.f;
    for (int i = threadIdx.x; i < DD; i += 256) { float v = xr[i] + yr[i]; ss += v * v; }
    red[threadIdx.x] = ss; __syncthreads();
    for (int s = 128; s > 0; s >>= 1) {
        if ((int)threadIdx.x < s) red[threadIdx.x] += red[threadIdx.x + s];
        __syncthreads();
    }
    float r = rsqrtf(red[0] * (1.f / DD) + 1e-5f);
    for (int i = threadIdx.x; i < DD; i += 256) {
        float v = (xr[i] + yr[i]) * r;
        outf[row * DD + i] = v;
        outb[row * DD + i] = f2bfbits(v);
    }
}

// ---------------------------------------------------------------------------
// act = silu(gate) * up, written bf16 (consumed only by the down GEMM)
// ---------------------------------------------------------------------------
__global__ __launch_bounds__(256)
void silu_mul(const float* __restrict__ gu, unsigned short* __restrict__ act) {
    long long idx = (long long)blockIdx.x * 256 + threadIdx.x;
    long long row = idx / II;
    int col = (int)(idx % II);
    float g = gu[row * (2 * II) + col];
    float u = gu[row * (2 * II) + II + col];
    act[idx] = f2bfbits(g / (1.f + __expf(-g)) * u);
}

// ---------------------------------------------------------------------------
// out (+)= w_b * rms_norm(h1 + mlp)
// ---------------------------------------------------------------------------
__global__ __launch_bounds__(256)
void final_accum(const float* __restrict__ h1, const float* __restrict__ mlp,
                 const float* __restrict__ expert_w, int s, int first,
                 float* __restrict__ out) {
    __shared__ float red[256];
    const size_t row = blockIdx.x;
    const int b = (int)(row >> 9);
    const float w = expert_w[b * EE + s];
    const float* xr = h1 + row * DD;
    const float* yr = mlp + row * DD;
    float ss = 0.f;
    for (int i = threadIdx.x; i < DD; i += 256) { float v = xr[i] + yr[i]; ss += v * v; }
    red[threadIdx.x] = ss; __syncthreads();
    for (int st = 128; st > 0; st >>= 1) {
        if ((int)threadIdx.x < st) red[threadIdx.x] += red[threadIdx.x + st];
        __syncthreads();
    }
    float r = rsqrtf(red[0] * (1.f / DD) + 1e-5f);
    for (int i = threadIdx.x; i < DD; i += 256) {
        float v = (xr[i] + yr[i]) * r * w;
        if (first) out[row * DD + i] = v;
        else       out[row * DD + i] += v;
    }
}

// ---------------------------------------------------------------------------
extern "C" void kernel_launch(void* const* d_in, const int* in_sizes, int n_in,
                              void* d_out, int out_size, void* d_ws, size_t ws_size,
                              hipStream_t stream) {
    const float* hidden      = (const float*)d_in[0];
    const float* cosT        = (const float*)d_in[1];
    const float* sinT        = (const float*)d_in[2];
    const float* router_w    = (const float*)d_in[3];
    const float* temperature = (const float*)d_in[4];
    const float* qkv_w       = (const float*)d_in[5];
    const float* o_w         = (const float*)d_in[6];
    const float* gu_w        = (const float*)d_in[7];
    const float* dn_w        = (const float*)d_in[8];

    float* out        = (float*)d_out;
    float* logits_out = out + (size_t)BT * DD;

    char* p = (char*)d_ws;
    float*          expert_w = (float*)p;           p += 4096;
    unsigned short* wbf      = (unsigned short*)p;  p += 8ull << 20;            // weights bf16
    unsigned short* hbf      = (unsigned short*)p;  p += (size_t)BT * DD * 2;   // hidden bf16
    unsigned short* abf      = (unsigned short*)p;  p += (size_t)BT * DD * 2;   // activations bf16
    float*          bufA     = (float*)p;           p += (size_t)BT * 3 * DD * 4; // qkv/oproj/gu/mlp
    float*          bufC     = (float*)p;                                        // h1 fp32

    const int attn_lds = 14336 + 64 * TT * 4;   // 145408 bytes
    const int gemm_lds = 16384;

    router_kernel<<<BB, 256, 0, stream>>>(hidden, router_w, temperature,
                                          logits_out, expert_w);
    f32_to_bf16_kernel<<<(BT * DD) / 256, 256, 0, stream>>>(hidden, hbf, (long long)BT * DD);

    for (int s = 0; s < EE; ++s) {
        const float* qw = qkv_w + (size_t)s * 3 * DD * DD;
        const float* ow = o_w   + (size_t)s * DD * DD;
        const float* gw = gu_w  + (size_t)s * 2 * II * DD;
        const float* dw = dn_w  + (size_t)s * DD * II;

        // qkv = h @ qkv_w^T -> bufA (fp32, BT x 3072)
        f32_to_bf16_kernel<<<(3 * DD * DD) / 256, 256, 0, stream>>>(qw, wbf, 3LL * DD * DD);
        gemm_bf16_wmma<<<dim3(BT / 64, (3 * DD) / 64), 128, gemm_lds, stream>>>(
            hbf, wbf, bufA, BT, 3 * DD, DD);
        rope_kernel<<<(BT * 2 * HH * 32) / 256, 256, 0, stream>>>(bufA, cosT, sinT);
        // attention -> abf (bf16, BT x 1024)
        attention_kernel<<<dim3(TT / 64, HH, BB), 128, attn_lds, stream>>>(bufA, abf);
        // o-proj -> bufA (fp32, BT x 1024); h1 = rms_norm(h + oproj) -> bufC + abf
        f32_to_bf16_kernel<<<(DD * DD) / 256, 256, 0, stream>>>(ow, wbf, (long long)DD * DD);
        gemm_bf16_wmma<<<dim3(BT / 64, DD / 64), 128, gemm_lds, stream>>>(
            abf, wbf, bufA, BT, DD, DD);
        rmsnorm_residual<<<BT, 256, 0, stream>>>(hidden, bufA, bufC, abf);
        // gu = h1 @ gate_up^T -> bufA (fp32, BT x 1536)
        f32_to_bf16_kernel<<<(2 * II * DD) / 256, 256, 0, stream>>>(gw, wbf, 2LL * II * DD);
        gemm_bf16_wmma<<<dim3(BT / 64, (2 * II) / 64), 128, gemm_lds, stream>>>(
            abf, wbf, bufA, BT, 2 * II, DD);
        // act = silu(gate)*up -> abf (bf16, BT x 768)
        silu_mul<<<(BT * II) / 256, 256, 0, stream>>>(bufA, abf);
        // mlp = act @ down^T -> bufA (fp32, BT x 1024)
        f32_to_bf16_kernel<<<(DD * II) / 256, 256, 0, stream>>>(dw, wbf, (long long)DD * II);
        gemm_bf16_wmma<<<dim3(BT / 64, DD / 64), 128, gemm_lds, stream>>>(
            abf, wbf, bufA, BT, DD, II);
        // result (+)= w * rms_norm(h1 + mlp)
        final_accum<<<BT, 256, 0, stream>>>(bufC, bufA, expert_w, s,
                                            (s == 0) ? 1 : 0, out);
    }
}
